// NeuronRouter_19035295056634
// MI455X (gfx1250) — matrix operand order repaired and checked by reading, not compiled, not run on hardware
//
#include <hip/hip_runtime.h>
#include <hip/hip_bf16.h>
#include <cstdint>
#include <cstddef>

#define DMODEL 512
#define NNEUR  2048
#define NHEADS 16
#define DHEAD  32
#define BB     8
#define SSEQ   1024
#define MTOK   (BB*SSEQ)   /* 8192 tokens */
#define TOPK   32

typedef __attribute__((ext_vector_type(16))) __bf16        v16bf;
typedef __attribute__((ext_vector_type(8)))  float         v8f;
typedef __attribute__((ext_vector_type(8)))  unsigned int  v8u;
typedef __attribute__((ext_vector_type(4)))  unsigned int  u32x4;
typedef __attribute__((ext_vector_type(4)))  int           i32x4;
typedef __attribute__((ext_vector_type(8)))  int           i32x8;

// ---------- helpers ----------
__device__ __forceinline__ unsigned short f2bf(float f) {
  unsigned int u = __float_as_uint(f);
  unsigned int r = u + 0x7FFFu + ((u >> 16) & 1u);   // round-to-nearest-even
  return (unsigned short)(r >> 16);
}

__device__ __forceinline__ v16bf mk16(uint4 a, uint4 b) {
  v8u u; u[0]=a.x; u[1]=a.y; u[2]=a.z; u[3]=a.w; u[4]=b.x; u[5]=b.y; u[6]=b.z; u[7]=b.w;
  return __builtin_bit_cast(v16bf, u);
}

// A-matrix (16x32 bf16) fragment: per ISA lanes0-15 hold K {0..7,16..23},
// lanes16-31 hold K {8..15,24..31} of row (lane&15). `row` = K-contiguous row ptr.
__device__ __forceinline__ v16bf loadA(const unsigned short* row, int k, int hi) {
  const uint4* p0 = reinterpret_cast<const uint4*>(row + k + (hi ? 8 : 0));
  const uint4* p1 = reinterpret_cast<const uint4*>(row + k + (hi ? 8 : 0) + 16);
  return mk16(p0[0], p1[0]);
}
// B-matrix (32x16 bf16) fragment: lanes0-15 hold K {0..7,16..23}, lanes16-31 K {8..15,24..31}
// of column (lane&15). `row` = K-contiguous column data (i.e. row of B^T).
__device__ __forceinline__ v16bf loadB(const unsigned short* row, int k, int hi) {
  const uint4* p0 = reinterpret_cast<const uint4*>(row + k + (hi ? 8 : 0));
  const uint4* p1 = reinterpret_cast<const uint4*>(row + k + (hi ? 8 : 0) + 16);
  return mk16(p0[0], p1[0]);
}

#define WMMA_BF16(a,b,c) __builtin_amdgcn_wmma_f32_16x16x32_bf16(false,(a),false,(b),(short)0,(c),false,false)

// LDS byte offset of a __shared__ pointer = low 32 bits of its flat address
__device__ __forceinline__ unsigned lds_offset(const void* p) {
  return (unsigned)(unsigned long long)(uintptr_t)p;
}

// ---------- TDM: DMA `rows` x `rowlen` bf16 tile (row-major, stride=rowlen) into LDS ----------
// D# per CDNA5 ISA ch.8: group0 = {count|..., lds_addr, global_addr, type=2};
// group1 = {data_size=1(2B), tensor_dim0/1, tile_dim0/1, dim0_stride}; groups 2/3 = 0 (2-D).
// Call from ONE wave only; waits TENSORcnt==0 before returning.
__device__ __forceinline__ void tdm_load_rows(const unsigned short* gsrc,
                                              unsigned short* ldst,
                                              int rows, int rowlen) {
#if defined(__HIP_DEVICE_COMPILE__) && __has_builtin(__builtin_amdgcn_tensor_load_to_lds)
  unsigned long long ga = (unsigned long long)gsrc;
  u32x4 g0;
  g0[0] = 1u;                                   // count=1, is_restore=0, gather=0
  g0[1] = lds_offset(ldst);                     // LDS byte address
  g0[2] = (unsigned)ga;                         // global_addr[31:0]
  g0[3] = (unsigned)(ga >> 32) | (2u << 30);    // global_addr[56:32] | type=2
  i32x8 g1;
  g1[0] = 1 << 16;                              // workgroup_mask=0, data_size=1 (2B)
  g1[1] = (int)((unsigned)rowlen << 16);        // tensor_dim0[15:0]  (bits 63:48)
  g1[2] = (int)((unsigned)rows   << 16);        // tensor_dim1[15:0]  (bits 95:80)
  g1[3] = (int)((unsigned)rowlen << 16);        // tile_dim0          (bits 127:112)
  g1[4] = rows;                                 // tile_dim1          (bits 143:128)
  g1[5] = rowlen;                               // tensor_dim0_stride (bits 191:160)
  g1[6] = 0;
  g1[7] = 0;
  i32x4 z4 = {};
#if defined(__clang_major__) && (__clang_major__ >= 23)
  i32x8 z8 = {};
  __builtin_amdgcn_tensor_load_to_lds(g0, g1, z4, z4, z8, 0);
#else
  __builtin_amdgcn_tensor_load_to_lds(g0, g1, z4, z4, 0);
#endif
  __builtin_amdgcn_s_wait_tensorcnt(0);
#else
  // fallback: cooperative copy by the calling wave (32 lanes, b128 chunks)
  const uint4* s = reinterpret_cast<const uint4*>(gsrc);
  uint4* d = reinterpret_cast<uint4*>(ldst);
  for (int i = (int)(threadIdx.x & 31); i < rows * rowlen / 8; i += 32) d[i] = s[i];
#endif
}

// ---------- K0: f32 -> bf16 conversion ----------
__global__ void cvt_bf16_kernel(const float* __restrict__ src,
                                unsigned short* __restrict__ dst, int n) {
  int i = blockIdx.x * 256 + threadIdx.x;
  if (i < n) dst[i] = f2bf(src[i]);
}

// ---------- K1: fused QKV projection  Y = X @ W^T + b ----------
// A tile (16 x 512 bf16) staged once per block via TDM; 4 waves share it.
// Q,K stored [B,H,S,dh] (dh contiguous); V stored transposed [B,H,dh,S].
__global__ __launch_bounds__(128) void proj_kernel(
    const unsigned short* __restrict__ xbf,
    const unsigned short* __restrict__ wq, const unsigned short* __restrict__ wk,
    const unsigned short* __restrict__ wv,
    const float* __restrict__ bq, const float* __restrict__ bk,
    const float* __restrict__ bv,
    unsigned short* __restrict__ qout, unsigned short* __restrict__ kout,
    unsigned short* __restrict__ vtout) {
  __shared__ __align__(16) unsigned short astage[16 * DMODEL];  // 16 KB
  const int wave = threadIdx.x >> 5, lane = threadIdx.x & 31;
  const int hi = lane >> 4, ln = lane & 15;
  const int mBase = blockIdx.x * 16;
  const int nBase = blockIdx.y * 64 + wave * 16;
  const int sel = blockIdx.z;
  const unsigned short* W = (sel == 0) ? wq : (sel == 1) ? wk : wv;
  const float* bias = (sel == 0) ? bq : (sel == 1) ? bk : bv;

  if (wave == 0)
    tdm_load_rows(xbf + (size_t)mBase * DMODEL, astage, 16, DMODEL);
  __syncthreads();

  const unsigned short* arow = astage + ln * DMODEL;
  const unsigned short* brow = W + (size_t)(nBase + ln) * DMODEL;
  v8f c = {};
#pragma unroll
  for (int ks = 0; ks < DMODEL / 32; ++ks) {
    v16bf a = loadA(arow, ks * 32, hi);
    v16bf b = loadB(brow, ks * 32, hi);
    c = WMMA_BF16(a, b, c);
  }
  const int n = nBase + ln;
  const float bs = bias[n];
  const int h = n >> 5, d = n & 31;
#pragma unroll
  for (int r = 0; r < 8; ++r) {
    int m = mBase + r + 8 * hi;
    int bb = m >> 10, s = m & (SSEQ - 1);
    unsigned short v = f2bf(c[r] + bs);
    if (sel == 0)      qout [((size_t)(bb * NHEADS + h) * SSEQ + s) * DHEAD + d] = v;
    else if (sel == 1) kout [((size_t)(bb * NHEADS + h) * SSEQ + s) * DHEAD + d] = v;
    else               vtout[((size_t)(bb * NHEADS + h) * DHEAD + d) * SSEQ + s] = v;
  }
}

// ---------- K2: flash attention, 1 wave per 16 query rows ----------
__global__ __launch_bounds__(128) void attn_kernel(
    const unsigned short* __restrict__ qbf, const unsigned short* __restrict__ kbf,
    const unsigned short* __restrict__ vt, float* __restrict__ ctx_out) {
  __shared__ __align__(16) unsigned short pstage[4][16 * 32];  // per-wave prob tile
  const int wave = threadIdx.x >> 5, lane = threadIdx.x & 31;
  const int hi = lane >> 4, ln = lane & 15;
  const int b = blockIdx.z, h = blockIdx.y;
  const int q0 = blockIdx.x * 64 + wave * 16;
  const size_t headQK = (size_t)(b * NHEADS + h) * SSEQ;
  const size_t headV  = (size_t)(b * NHEADS + h) * DHEAD;

  v16bf aQ = loadA(qbf + (headQK + q0 + ln) * DHEAD, 0, hi);  // invariant A

  v8f c0 = {}, c1 = {};
  float rm[8], rl[8];
#pragma unroll
  for (int r = 0; r < 8; ++r) { rm[r] = -3.0e38f; rl[r] = 0.f; }
  const float scale = 0.17677669529663687f;   // 1/sqrt(32)
  unsigned short* pb = pstage[wave];

  for (int kc = 0; kc < SSEQ / 32; ++kc) {
    const int k0 = kc * 32;
    v16bf bK0 = loadB(kbf + (headQK + k0 + ln) * DHEAD, 0, hi);
    v16bf bK1 = loadB(kbf + (headQK + k0 + 16 + ln) * DHEAD, 0, hi);
    v8f s0 = {}, s1 = {};
    s0 = WMMA_BF16(aQ, bK0, s0);
    s1 = WMMA_BF16(aQ, bK1, s1);
#pragma unroll
    for (int r = 0; r < 8; ++r) {
      float a0 = s0[r] * scale, a1 = s1[r] * scale;
      float tm = fmaxf(a0, a1);
#pragma unroll
      for (int m = 1; m < 16; m <<= 1) tm = fmaxf(tm, __shfl_xor(tm, m, 32));
      float mnew = fmaxf(rm[r], tm);
      float p0 = __expf(a0 - mnew), p1 = __expf(a1 - mnew);
      float ts = p0 + p1;
#pragma unroll
      for (int m = 1; m < 16; m <<= 1) ts += __shfl_xor(ts, m, 32);
      float corr = __expf(rm[r] - mnew);
      rl[r] = rl[r] * corr + ts;
      rm[r] = mnew;
      c0[r] *= corr; c1[r] *= corr;
      int mrow = r + 8 * hi;                       // C-layout -> (m,kk) LDS transpose
      pb[mrow * 32 + ln]      = f2bf(p0);
      pb[mrow * 32 + 16 + ln] = f2bf(p1);
    }
    asm volatile("s_wait_dscnt 0" ::: "memory");   // probs visible before A read
    v16bf aP  = loadA(pb + ln * 32, 0, hi);
    v16bf bV0 = loadB(vt + (headV + ln) * SSEQ, k0, hi);
    v16bf bV1 = loadB(vt + (headV + 16 + ln) * SSEQ, k0, hi);
    c0 = WMMA_BF16(aP, bV0, c0);
    c1 = WMMA_BF16(aP, bV1, c1);
  }
#pragma unroll
  for (int r = 0; r < 8; ++r) {
    int mrow = r + 8 * hi;
    int s = q0 + mrow;
    float inv = 1.0f / rl[r];
    size_t base = ((size_t)(b * SSEQ + s)) * DMODEL + h * DHEAD;
    ctx_out[base + ln]      = c0[r] * inv;
    ctx_out[base + 16 + ln] = c1[r] * inv;
  }
}

// ---------- K3a: gate weights + mixed vector m = w0*x + w1*ctx (bf16) ----------
__global__ __launch_bounds__(128) void mix_kernel(
    const float* __restrict__ x, const float* __restrict__ ctx,
    const float* __restrict__ pw, const float* __restrict__ pbv,
    unsigned short* __restrict__ mbf) {
  __shared__ float r0[128], r1[128];
  const int tok = blockIdx.x, t = threadIdx.x;
  const float* xr = x   + (size_t)tok * DMODEL;
  const float* cr = ctx + (size_t)tok * DMODEL;
  float a0 = 0.f, a1 = 0.f;
  for (int i = t; i < DMODEL; i += 128) {
    float xv = xr[i], cv = cr[i];
    a0 += xv * pw[i]              + cv * pw[DMODEL + i];
    a1 += xv * pw[2 * DMODEL + i] + cv * pw[3 * DMODEL + i];
  }
  r0[t] = a0; r1[t] = a1;
  __syncthreads();
  for (int s = 64; s > 0; s >>= 1) {
    if (t < s) { r0[t] += r0[t + s]; r1[t] += r1[t + s]; }
    __syncthreads();
  }
  float s0 = r0[0] + pbv[0], s1 = r1[0] + pbv[1];
  float mx = fmaxf(s0, s1);
  float e0 = __expf(s0 - mx), e1 = __expf(s1 - mx);
  float w0 = e0 / (e0 + e1), w1 = e1 / (e0 + e1);
  for (int i = t; i < DMODEL; i += 128)
    mbf[(size_t)tok * DMODEL + i] = f2bf(w0 * xr[i] + w1 * cr[i]);
}

// ---------- K3b: blended scores = m @ neurons^T (WMMA, A staged via TDM) ----------
__global__ __launch_bounds__(128) void score_kernel(
    const unsigned short* __restrict__ mbf, const unsigned short* __restrict__ nbf,
    float* __restrict__ scores) {
  __shared__ __align__(16) unsigned short astage[16 * DMODEL];  // 16 KB
  const int wave = threadIdx.x >> 5, lane = threadIdx.x & 31;
  const int hi = lane >> 4, ln = lane & 15;
  const int mBase = blockIdx.x * 16;
  const int nBase = blockIdx.y * 64 + wave * 16;

  if (wave == 0)
    tdm_load_rows(mbf + (size_t)mBase * DMODEL, astage, 16, DMODEL);
  __syncthreads();

  const unsigned short* arow = astage + ln * DMODEL;
  const unsigned short* brow = nbf + (size_t)(nBase + ln) * DMODEL;
  v8f c = {};
#pragma unroll
  for (int ks = 0; ks < DMODEL / 32; ++ks) {
    v16bf a = loadA(arow, ks * 32, hi);
    v16bf b = loadB(brow, ks * 32, hi);
    c = WMMA_BF16(a, b, c);
  }
#pragma unroll
  for (int r = 0; r < 8; ++r) {
    int m = mBase + r + 8 * hi;
    scores[(size_t)m * NNEUR + nBase + ln] = c[r];
  }
}

// ---------- K3c: per-token top-32 + softmax + 512MB gather ----------
__global__ __launch_bounds__(128) void topk_kernel(
    const float* __restrict__ scores, const float* __restrict__ neurons,
    float* __restrict__ selected, float* __restrict__ idxOut,
    float* __restrict__ wOut) {
  __shared__ float row[NNEUR];   // 8 KB, L2-resident source
  __shared__ int sidx[TOPK];
  const int tok = blockIdx.x, t = threadIdx.x;
  for (int i = t; i < NNEUR; i += 128) row[i] = scores[(size_t)tok * NNEUR + i];
  __syncthreads();
  if (t < 32) {
    const int lane = t;
    float myv = -3.0e38f; int myi = 0;
    for (int it = 0; it < TOPK; ++it) {
      float v = -3.0e38f; int ix = 0;
      for (int i = lane; i < NNEUR; i += 32) {
        float xv = row[i];
        if (xv > v) { v = xv; ix = i; }
      }
#pragma unroll
      for (int m = 16; m >= 1; m >>= 1) {      // argmax, ties -> lowest index
        float ov = __shfl_xor(v, m, 32);
        int   oi = __shfl_xor(ix, m, 32);
        if (ov > v || (ov == v && oi < ix)) { v = ov; ix = oi; }
      }
      if (lane == it) { myv = v; myi = ix; }
      if (lane == 0) row[ix] = -3.0e38f;
      asm volatile("s_wait_dscnt 0" ::: "memory");
    }
    float mx = myv;
#pragma unroll
    for (int m = 16; m >= 1; m >>= 1) mx = fmaxf(mx, __shfl_xor(mx, m, 32));
    float e = __expf(myv - mx);
    float ssum = e;
#pragma unroll
    for (int m = 16; m >= 1; m >>= 1) ssum += __shfl_xor(ssum, m, 32);
    idxOut[(size_t)tok * TOPK + lane] = (float)myi;  // int idx as f32 (flat f32 d_out)
    wOut  [(size_t)tok * TOPK + lane] = e / ssum;
    sidx[lane] = myi;
  }
  __syncthreads();
  for (int kk = 0; kk < TOPK; ++kk) {
    int ni = sidx[kk];
    if (kk + 1 < TOPK) __builtin_prefetch(neurons + (size_t)sidx[kk + 1] * DMODEL, 0, 1);
    const float4* src = reinterpret_cast<const float4*>(neurons + (size_t)ni * DMODEL);
    float4* dst = reinterpret_cast<float4*>(selected + ((size_t)tok * TOPK + kk) * DMODEL);
    dst[t] = src[t];                               // 128 lanes * float4 = 512 floats
  }
}

// ---------- host ----------
extern "C" void kernel_launch(void* const* d_in, const int* in_sizes, int n_in,
                              void* d_out, int out_size, void* d_ws, size_t ws_size,
                              hipStream_t stream) {
  (void)in_sizes; (void)n_in; (void)out_size; (void)ws_size;
  const float* x  = (const float*)d_in[0];
  const float* nr = (const float*)d_in[1];
  const float* qw = (const float*)d_in[2];
  const float* qb = (const float*)d_in[3];
  const float* kw = (const float*)d_in[4];
  const float* kb = (const float*)d_in[5];
  const float* vw = (const float*)d_in[6];
  const float* vb = (const float*)d_in[7];
  const float* pw = (const float*)d_in[8];
  const float* pb = (const float*)d_in[9];

  float* out = (float*)d_out;
  const size_t SEL_SZ = (size_t)MTOK * TOPK * DMODEL;  // 134217728
  const size_t IDX_SZ = (size_t)MTOK * TOPK;           // 262144
  float* selOut = out;
  float* idxOut = out + SEL_SZ;
  float* wOut   = idxOut + IDX_SZ;
  float* ctxOut = wOut + IDX_SZ;                       // 4194304 f32

  unsigned short* xbf  = (unsigned short*)d_ws;
  unsigned short* qwbf = xbf  + (size_t)MTOK * DMODEL;
  unsigned short* kwbf = qwbf + DMODEL * DMODEL;
  unsigned short* vwbf = kwbf + DMODEL * DMODEL;
  unsigned short* nbf  = vwbf + DMODEL * DMODEL;
  unsigned short* qbf  = nbf  + (size_t)NNEUR * DMODEL;
  unsigned short* kbf  = qbf  + (size_t)MTOK * DMODEL;
  unsigned short* vt   = kbf  + (size_t)MTOK * DMODEL;
  unsigned short* mbf  = vt   + (size_t)MTOK * DMODEL;
  float* scores = (float*)(mbf + (size_t)MTOK * DMODEL);  // 64MB, L2-resident

  cvt_bf16_kernel<<<(MTOK * DMODEL + 255) / 256, 256, 0, stream>>>(x, xbf, MTOK * DMODEL);
  cvt_bf16_kernel<<<(DMODEL * DMODEL + 255) / 256, 256, 0, stream>>>(qw, qwbf, DMODEL * DMODEL);
  cvt_bf16_kernel<<<(DMODEL * DMODEL + 255) / 256, 256, 0, stream>>>(kw, kwbf, DMODEL * DMODEL);
  cvt_bf16_kernel<<<(DMODEL * DMODEL + 255) / 256, 256, 0, stream>>>(vw, vwbf, DMODEL * DMODEL);
  cvt_bf16_kernel<<<(NNEUR * DMODEL + 255) / 256, 256, 0, stream>>>(nr, nbf, NNEUR * DMODEL);

  proj_kernel<<<dim3(MTOK / 16, DMODEL / 64, 3), 128, 0, stream>>>(
      xbf, qwbf, kwbf, vwbf, qb, kb, vb, qbf, kbf, vt);

  attn_kernel<<<dim3(SSEQ / 64, NHEADS, BB), 128, 0, stream>>>(qbf, kbf, vt, ctxOut);

  mix_kernel<<<MTOK, 128, 0, stream>>>(x, ctxOut, pw, pb, mbf);

  score_kernel<<<dim3(MTOK / 16, NNEUR / 64), 128, 0, stream>>>(mbf, nbf, scores);

  topk_kernel<<<MTOK, 128, 0, stream>>>(scores, nr, selOut, idxOut, wOut);
}
